// LayerNormGRU_10419590660742
// MI455X (gfx1250) — matrix operand order repaired and checked by reading, not compiled, not run on hardware
//
#include <hip/hip_runtime.h>

#define LSEQ 512
#define BATCH 64
#define IN 512
#define HID 512
#define G3 1536   // 3*HID
#define LN_EPS 1e-5f

typedef __attribute__((ext_vector_type(16))) __bf16 v16bf;
typedef __attribute__((ext_vector_type(8)))  __bf16 v8bf;
typedef __attribute__((ext_vector_type(8)))  float  v8f;

// ---------------------------------------------------------------------------
// WMMA fragment loaders (16x16x32 bf16; A/B use this code's contiguous k order)
// A (16xK, 16-bit): lanes 0-15 -> rows M=0..15, elems 0..15 = K k0..k0+15 ;
//                   lanes 16-31: k0 += 16.
// B (Kx16, 16-bit): lanes 0-15 -> cols N, K = kb..kb+15 ; lanes 16-31 K += 16.
// C (16x16 f32):    lane l, vgpr v -> (M = v + 8*(l>=16), N = l%16).
// ---------------------------------------------------------------------------
__device__ __forceinline__ v16bf load_a_frag_lds(const __bf16* a, int lane,
                                                 int kb, int kstride) {
  int row = lane & 15;
  int k0  = kb + ((lane & 16) ? 16 : 0);
  const __bf16* base = a + row * kstride + k0;
  v8bf lo = *(const v8bf*)(base);
  v8bf hi = *(const v8bf*)(base + 8);
  return __builtin_shufflevector(lo, hi, 0,1,2,3,4,5,6,7,8,9,10,11,12,13,14,15);
}

__device__ __forceinline__ v16bf load_b_frag_g(const __bf16* __restrict__ B, int lane,
                                               int n0, int kb, int kstride) {
  int col = n0 + (lane & 15);
  int k0  = kb + ((lane & 16) ? 16 : 0);
  return *(const v16bf*)(B + (size_t)col * kstride + k0);
}

template<int NT>
__device__ __forceinline__ void load_b_set(const __bf16* __restrict__ gB, int lane,
                                           int n_base, int kb, v16bf b[NT]) {
#pragma unroll
  for (int nt = 0; nt < NT; ++nt) b[nt] = load_b_frag_g(gB, lane, n_base + nt * 16, kb, 512);
}

// Per-wave pipelined GEMM: acc[16 x (NT*16)] = A[16 x 512] * B^T.
// B fragments are double-buffered in registers so the next K-slab's global
// loads are in flight while the current slab's WMMAs execute.
template<int NT>
__device__ __forceinline__ void gemm_rowtile(const __bf16* lds_a,
                                             const __bf16* __restrict__ gB,
                                             int lane, int n_base, v8f acc[NT]) {
  const v8f vzero = {0.f,0.f,0.f,0.f,0.f,0.f,0.f,0.f};
#pragma unroll
  for (int i = 0; i < NT; ++i) acc[i] = vzero;

  v16bf b0[NT], b1[NT];
  load_b_set<NT>(gB, lane, n_base, 0, b0);

  for (int kb = 0; kb < 512; kb += 64) {
    load_b_set<NT>(gB, lane, n_base, kb + 32, b1);   // in flight during b0 WMMAs
    v16bf a0 = load_a_frag_lds(lds_a, lane, kb, 512);
#pragma unroll
    for (int nt = 0; nt < NT; ++nt)
      acc[nt] = __builtin_amdgcn_wmma_f32_16x16x32_bf16(
          false, a0, false, b0[nt], (short)0, acc[nt], false, false);

    if (kb + 64 < 512) load_b_set<NT>(gB, lane, n_base, kb + 64, b0);
    v16bf a1 = load_a_frag_lds(lds_a, lane, kb + 32, 512);
#pragma unroll
    for (int nt = 0; nt < NT; ++nt)
      acc[nt] = __builtin_amdgcn_wmma_f32_16x16x32_bf16(
          false, a1, false, b1[nt], (short)0, acc[nt], false, false);
  }
}

// Scatter C fragments (+bias) into an LDS [16 x G3] fp32 tile.
template<int NT>
__device__ __forceinline__ void store_acc(float* tile, const float* __restrict__ bias,
                                          int lane, int n_base, const v8f acc[NT]) {
  int rbase = (lane & 16) ? 8 : 0;
  int cl = lane & 15;
#pragma unroll
  for (int nt = 0; nt < NT; ++nt) {
    int col = n_base + nt * 16 + cl;
    float bv = bias[col];
#pragma unroll
    for (int v = 0; v < 8; ++v) {
      tile[(rbase + v) * G3 + col] = acc[nt][v] + bv;
    }
  }
}

// Row LayerNorm of a [16 x G3] LDS tile, in place. 256 threads: 16 per row.
// red: 512-float LDS scratch. Caller must __syncthreads() before and after.
__device__ __forceinline__ void layernorm16(float* tile, const float* __restrict__ g,
                                            const float* __restrict__ bta,
                                            float* red, int tid) {
  const int CPT = G3 / 16;  // 96 columns per thread
  int row = tid >> 4, sub = tid & 15;
  float* p = tile + row * G3 + sub * CPT;
  float s = 0.f, ss = 0.f;
#pragma unroll 4
  for (int i = 0; i < CPT; ++i) { float v = p[i]; s += v; ss += v * v; }
  red[row * 16 + sub]       = s;
  red[256 + row * 16 + sub] = ss;
  __syncthreads();
  float mu = 0.f, m2 = 0.f;
#pragma unroll
  for (int i = 0; i < 16; ++i) { mu += red[row * 16 + i]; m2 += red[256 + row * 16 + i]; }
  mu *= (1.f / G3);
  m2 = m2 * (1.f / G3) - mu * mu;
  float rstd = rsqrtf(m2 + LN_EPS);
#pragma unroll 4
  for (int i = 0; i < CPT; ++i) {
    int col = sub * CPT + i;
    p[i] = (p[i] - mu) * rstd * g[col] + bta[col];
  }
}

// ---------------------------------------------------------------------------
// Kernel 0: one-shot fp32 -> bf16 conversion of both weight matrices.
// ---------------------------------------------------------------------------
__global__ void k_cvt_bf16(const float* __restrict__ a, const float* __restrict__ b,
                           __bf16* __restrict__ oa, __bf16* __restrict__ ob, int n) {
  int i = blockIdx.x * blockDim.x + threadIdx.x;
  if (i < n) { oa[i] = (__bf16)a[i]; ob[i] = (__bf16)b[i]; }
}

// ---------------------------------------------------------------------------
// Kernel 1: gi_all = LN(x @ w_ih^T + b_ih)  for all L*B rows.
// Grid = 2048 WGs, each WG computes one [16 x 1536] output row-tile.
// ---------------------------------------------------------------------------
__global__ __launch_bounds__(256) void k_gates_ih(
    const float* __restrict__ x, const __bf16* __restrict__ wih_bf,
    const float* __restrict__ b_ih, const float* __restrict__ g_ih,
    const float* __restrict__ beta_ih, float* __restrict__ gi_all) {
  __shared__ __align__(16) __bf16 lds_a[16 * IN];
  __shared__ __align__(16) float  tile[16 * G3];
  __shared__ float red[512];
  int tid = threadIdx.x;
  size_t row0 = (size_t)blockIdx.x * 16;

  const float* xb = x + row0 * IN;
  for (int i = tid; i < 16 * IN; i += 256) lds_a[i] = (__bf16)xb[i];
  __syncthreads();

  int lane = tid & 31, wave = tid >> 5;
  v8f acc[6];
  gemm_rowtile<6>(lds_a, wih_bf, lane, wave * 192, acc);
  store_acc<6>(tile, b_ih, lane, wave * 192, acc);
  gemm_rowtile<6>(lds_a, wih_bf, lane, wave * 192 + 96, acc);
  store_acc<6>(tile, b_ih, lane, wave * 192 + 96, acc);
  __syncthreads();
  layernorm16(tile, g_ih, beta_ih, red, tid);
  __syncthreads();

  float* out = gi_all + row0 * G3;
  for (int i = tid; i < 16 * G3; i += 256) out[i] = tile[i];
}

// ---------------------------------------------------------------------------
// Kernel 2: persistent recurrent scan. Batch rows are independent across the
// whole recurrence, so each WG owns 16 batch rows and runs all 512 steps with
// h resident in LDS and w_hh streamed from L2 — no inter-WG sync at all.
// Grid = 4 WGs x 256 threads.
// ---------------------------------------------------------------------------
__global__ __launch_bounds__(256) void k_recurrent(
    const float* __restrict__ hx, const __bf16* __restrict__ whh_bf,
    const float* __restrict__ b_hh, const float* __restrict__ g_hh,
    const float* __restrict__ beta_hh, const float* __restrict__ gi_all,
    float* __restrict__ out) {
  __shared__ __align__(16) float  h_f32[16 * HID];
  __shared__ __align__(16) __bf16 h_bf[16 * HID];
  __shared__ __align__(16) float  tile[16 * G3];
  __shared__ float red[512];

  int tid = threadIdx.x;
  int wg  = blockIdx.x;                 // owns batch rows 16*wg .. 16*wg+15
  int lane = tid & 31, wave = tid >> 5;

  const float* hx0 = hx + (size_t)wg * 16 * HID;
  for (int i = tid; i < 16 * HID; i += 256) {
    float v = hx0[i];
    h_f32[i] = v;
    h_bf[i]  = (__bf16)v;
  }
  __syncthreads();

  const size_t out_last = (size_t)LSEQ * BATCH * (2 * HID);

  for (int t = 0; t < LSEQ; ++t) {
    // Warm L2/L0 with this step's input-gate tile while the GEMM runs.
    const float* gi = gi_all + ((size_t)t * BATCH + (size_t)wg * 16) * G3;
#pragma unroll
    for (int pf = 0; pf < 3; ++pf)
      __builtin_prefetch(gi + ((size_t)tid * 3 + pf) * 32, 0, 1);

    // gh = LN(h @ w_hh^T + b_hh)
    v8f acc[6];
    gemm_rowtile<6>(h_bf, whh_bf, lane, wave * 192, acc);
    store_acc<6>(tile, b_hh, lane, wave * 192, acc);
    gemm_rowtile<6>(h_bf, whh_bf, lane, wave * 192 + 96, acc);
    store_acc<6>(tile, b_hh, lane, wave * 192 + 96, acc);
    __syncthreads();
    layernorm16(tile, g_hh, beta_hh, red, tid);
    __syncthreads();

    // Gates + state update + output writes (fwd slot and mirrored bwd slot).
    for (int idx = tid; idx < 16 * HID; idx += 256) {
      int m = idx >> 9;            // local batch row 0..15
      int j = idx & (HID - 1);     // hidden unit
      const float* gr = gi + (size_t)m * G3;
      float i_r = gr[j], i_i = gr[HID + j], i_n = gr[2 * HID + j];
      float h_r = tile[m * G3 + j];
      float h_i = tile[m * G3 + HID + j];
      float h_n = tile[m * G3 + 2 * HID + j];
      float r = 1.f / (1.f + __expf(-(i_r + h_r)));
      float z = 1.f / (1.f + __expf(-(i_i + h_i)));
      float n = tanhf(i_n + r * h_n);
      float hold = h_f32[idx];
      float hy = n + z * (hold - n);
      h_f32[idx] = hy;
      h_bf[idx]  = (__bf16)hy;

      int brow = wg * 16 + m;
      out[((size_t)t * BATCH + brow) * (2 * HID) + j] = hy;                       // fwd
      out[((size_t)(LSEQ - 1 - t) * BATCH + brow) * (2 * HID) + HID + j] = hy;    // bwd mirror
      if (t == LSEQ - 1) out[out_last + (size_t)brow * (2 * HID) + j] = hy;       // last fwd
      if (t == 0)        out[out_last + (size_t)brow * (2 * HID) + HID + j] = hy; // last bwd
    }
    __syncthreads();
  }
}

// ---------------------------------------------------------------------------
extern "C" void kernel_launch(void* const* d_in, const int* in_sizes, int n_in,
                              void* d_out, int out_size, void* d_ws, size_t ws_size,
                              hipStream_t stream) {
  const float* x       = (const float*)d_in[0];
  const float* hx      = (const float*)d_in[1];
  const float* w_ih    = (const float*)d_in[2];
  const float* b_ih    = (const float*)d_in[3];
  const float* w_hh    = (const float*)d_in[4];
  const float* b_hh    = (const float*)d_in[5];
  const float* g_ih    = (const float*)d_in[6];
  const float* beta_ih = (const float*)d_in[7];
  const float* g_hh    = (const float*)d_in[8];
  const float* beta_hh = (const float*)d_in[9];
  float* out = (float*)d_out;

  // Workspace layout
  char* ws = (char*)d_ws;
  const size_t wbytes = (size_t)G3 * IN * sizeof(__bf16);  // 1.5 MB each
  __bf16* wih_bf = (__bf16*)ws;
  __bf16* whh_bf = (__bf16*)(ws + wbytes);
  float*  gi_all = (float*)(ws + 2 * wbytes);              // [L*B, 3H] fp32

  const int nw = G3 * IN;
  hipLaunchKernelGGL(k_cvt_bf16, dim3((nw + 255) / 256), dim3(256), 0, stream,
                     w_ih, w_hh, wih_bf, whh_bf, nw);
  hipLaunchKernelGGL(k_gates_ih, dim3(LSEQ * BATCH / 16), dim3(256), 0, stream,
                     x, wih_bf, b_ih, g_ih, beta_ih, gi_all);
  hipLaunchKernelGGL(k_recurrent, dim3(BATCH / 16), dim3(256), 0, stream,
                     hx, whh_bf, b_hh, g_hh, beta_hh, gi_all, out);
}
